// Encoder_49177375539705
// MI455X (gfx1250) — matrix operand order repaired and checked by reading, not compiled, not run on hardware
//
#include <hip/hip_runtime.h>
#include <hip/hip_bf16.h>
#include <math.h>

// ---------------- problem constants ----------------
#define BATCH 4
#define SEQN  1024
#define SEQM  256
#define DMODEL 1024
#define NHEAD 16
#define DHEAD 64
#define NLAYER 6
#define NBUCKET 32

typedef __bf16 bf16_t;
typedef bf16_t bf16x16 __attribute__((ext_vector_type(16)));
typedef float  f32x8   __attribute__((ext_vector_type(8)));
typedef int    vi4     __attribute__((__vector_size__(16)));   // matches builtin proto

union Frag { bf16x16 v; unsigned u[8]; };

__device__ inline f32x8 wmma_bf16(const Frag& a, const Frag& b, f32x8 c) {
    return __builtin_amdgcn_wmma_f32_16x16x32_bf16(
        false, a.v, false, b.v, (short)0, c, false, false);
}

// ---------------- CDNA5 async global->LDS staging ----------------
#if __has_builtin(__builtin_amdgcn_global_load_async_to_lds_b128)
#define HAVE_ASYNC_LDS 1
#endif

__device__ inline void async_copy16(const bf16_t* g, bf16_t* l) {
#ifdef HAVE_ASYNC_LDS
    __builtin_amdgcn_global_load_async_to_lds_b128(
        (__attribute__((address_space(1))) vi4*)g,
        (__attribute__((address_space(3))) vi4*)l, 0, 0);
#else
    *(int4*)l = *(const int4*)g;
#endif
}

__device__ inline void wait_async_0() {
#if __has_builtin(__builtin_amdgcn_s_wait_asynccnt)
    __builtin_amdgcn_s_wait_asynccnt(0);
#else
    asm volatile("s_wait_asynccnt 0x0" ::: "memory");
#endif
}
__device__ inline void wait_async_4() {
#if __has_builtin(__builtin_amdgcn_s_wait_asynccnt)
    __builtin_amdgcn_s_wait_asynccnt(4);
#else
    asm volatile("s_wait_asynccnt 0x4" ::: "memory");
#endif
}

// A fragment: 16 rows x 32 k.  row = lane%16 ; half = lane/16.
__device__ inline void load_a_frag(Frag& f, const unsigned* rowPtr, int basePair, int half) {
#pragma unroll
    for (int g = 0; g < 8; ++g) {
        int p = basePair + 8 * half + g;
        f.u[g] = rowPtr[p];
    }
}
// B fragment: 32 k x 16 cols, column-major pairs (k contiguous per col).
__device__ inline void load_b_frag(Frag& f, const unsigned* colPtr, int basePair, int half) {
#pragma unroll
    for (int g = 0; g < 8; ++g) f.u[g] = colPtr[basePair + 8 * half + g];
}

// ---------------- weight convert + transpose: fp32 [K][N] -> bf16 [N][K] ----------------
__global__ __launch_bounds__(256)
void enc_convw_kernel(const float* __restrict__ W, int K, int N, bf16_t* __restrict__ WT) {
    __shared__ float t[32][33];
    int k0 = blockIdx.y * 32, n0 = blockIdx.x * 32;
    int c = threadIdx.x & 31, r8 = threadIdx.x >> 5;
#pragma unroll
    for (int i = 0; i < 4; ++i) {
        int r = r8 + 8 * i;
        t[r][c] = W[(size_t)(k0 + r) * N + n0 + c];
    }
    __syncthreads();
#pragma unroll
    for (int i = 0; i < 4; ++i) {
        int r = r8 + 8 * i;
        WT[(size_t)(n0 + r) * K + k0 + c] = (bf16_t)t[c][r];
    }
}

// ---------------- fp32 -> bf16 flat convert ----------------
__global__ void enc_convx_kernel(const float* __restrict__ a, bf16_t* __restrict__ b, int n) {
    int i = blockIdx.x * 256 + threadIdx.x;
    if (i < n) b[i] = (bf16_t)a[i];
}

// ---------------- T5 rel-pos bucket table ----------------
__global__ void enc_bucket_kernel(int* __restrict__ bucket1d, int nq, int nkv) {
    int idx = blockIdx.x * 256 + threadIdx.x;
    if (idx >= nq + nkv - 1) return;
    int rel = idx - (nq - 1);
    int n = rel < 0 ? -rel : 0;
    int bkt;
    if (n < 16) bkt = n;
    else {
        int vl = 16 + (int)(logf((float)n / 16.0f) / logf(8.0f) * 16.0f);
        bkt = vl < 31 ? vl : 31;
    }
    bucket1d[idx] = bkt;
}

// ---------------- layernorm: fp32 in -> bf16 out ----------------
__global__ __launch_bounds__(256)
void enc_layernorm_kernel(const float* __restrict__ x,
                          const float* __restrict__ g,
                          const float* __restrict__ b,
                          bf16_t* __restrict__ y) {
    int row = blockIdx.x;
    const float* xr = x + (size_t)row * DMODEL;
    int tid = threadIdx.x;
    float vals[4];
    float s = 0.f, s2 = 0.f;
#pragma unroll
    for (int i = 0; i < 4; ++i) {
        float v = xr[tid + 256 * i];
        vals[i] = v; s += v; s2 += v * v;
    }
    __shared__ float red[16];
    for (int o = 16; o; o >>= 1) { s += __shfl_xor(s, o, 32); s2 += __shfl_xor(s2, o, 32); }
    int wave = tid >> 5, lane = tid & 31;
    if (lane == 0) { red[wave] = s; red[8 + wave] = s2; }
    __syncthreads();
    float ts = 0.f, ts2 = 0.f;
#pragma unroll
    for (int w = 0; w < 8; ++w) { ts += red[w]; ts2 += red[8 + w]; }
    float mu = ts * (1.0f / DMODEL);
    float var = ts2 * (1.0f / DMODEL) - mu * mu;
    float rs = rsqrtf(var + 1e-5f);
#pragma unroll
    for (int i = 0; i < 4; ++i) {
        int c = tid + 256 * i;
        y[(size_t)row * DMODEL + c] = (bf16_t)((vals[i] - mu) * rs * g[c] + b[c]);
    }
}

// ---------------- WMMA GEMM (bf16 x bf16^T), double-buffered async staging ----------------
// A: bf16 [M][K] (lda), WT: bf16 [N][K] (ldwt). WG tile 128x128, 8 waves x (32x64).
__global__ __launch_bounds__(256)
void enc_gemm_kernel(const bf16_t* __restrict__ A, int lda,
                     const bf16_t* __restrict__ WT, int ldwt,
                     const float* __restrict__ bias,
                     const float* __restrict__ resid,
                     float* __restrict__ Cf, bf16_t* __restrict__ Ch, int ldc,
                     int K, int fuse_gelu) {
    __shared__ bf16_t lsA[2][128][32];
    __shared__ bf16_t lsB[2][128][32];
    int tid = threadIdx.x;
    int lane = tid & 31, wave = tid >> 5;
    int half = lane >> 4, l16 = lane & 15;
    int m0 = blockIdx.y * 128;
    int n0 = blockIdx.x * 128;
    int wr = (wave >> 1) * 32;
    int wc = (wave & 1) * 64;
    f32x8 acc[2][4];
#pragma unroll
    for (int i = 0; i < 2; ++i)
#pragma unroll
        for (int j = 0; j < 4; ++j) acc[i][j] = (f32x8)0.0f;

    auto stage = [&](int k0, int buf) {
        for (int c = tid; c < 512; c += 256) {        // A: 128 rows x 4 x 16B
            int r = c >> 2, q = c & 3;
            async_copy16(A + (size_t)(m0 + r) * lda + k0 + q * 8, &lsA[buf][r][q * 8]);
        }
        for (int c = tid; c < 512; c += 256) {        // B: 128 cols x 4 x 16B
            int r = c >> 2, q = c & 3;
            async_copy16(WT + (size_t)(n0 + r) * ldwt + k0 + q * 8, &lsB[buf][r][q * 8]);
        }
    };

    int nk = K >> 5;
    stage(0, 0);
    for (int k = 0; k < nk; ++k) {
        int buf = k & 1;
        if (k + 1 < nk) {          // overlap: k+1 copies in flight during WMMA on k
            stage((k + 1) << 5, buf ^ 1);
            wait_async_4();        // own stage-k copies done (async loads complete in order)
        } else {
            wait_async_0();
        }
        __syncthreads();           // all waves' stage-k data visible
        Frag a[2], b[4];
#pragma unroll
        for (int i = 0; i < 2; ++i)
            load_a_frag(a[i], (const unsigned*)&lsA[buf][wr + 16 * i + l16][0], 0, half);
#pragma unroll
        for (int j = 0; j < 4; ++j)
            load_b_frag(b[j], (const unsigned*)&lsB[buf][wc + 16 * j + l16][0], 0, half);
#pragma unroll
        for (int i = 0; i < 2; ++i)
#pragma unroll
            for (int j = 0; j < 4; ++j) acc[i][j] = wmma_bf16(a[i], b[j], acc[i][j]);
        __syncthreads();           // all waves done reading buf before re-staging it
    }
#pragma unroll
    for (int i = 0; i < 2; ++i)
#pragma unroll
        for (int j = 0; j < 4; ++j) {
            int col = n0 + wc + 16 * j + l16;
            float bv = bias ? bias[col] : 0.0f;
#pragma unroll
            for (int v = 0; v < 8; ++v) {
                int row = m0 + wr + 16 * i + v + 8 * half;
                float x = acc[i][j][v] + bv;
                if (fuse_gelu) x = 0.5f * x * (1.0f + erff(x * 0.70710678f));
                if (resid) x += resid[(size_t)row * ldc + col];
                if (Cf) Cf[(size_t)row * ldc + col] = x;
                else    Ch[(size_t)row * ldc + col] = (bf16_t)x;
            }
        }
}

// ---------------- attention scores: S = scale*Q@K^T (+bias)(+mask) -> bf16 ----------------
__global__ __launch_bounds__(256)
void enc_scores_kernel(const bf16_t* __restrict__ Q,
                       const bf16_t* __restrict__ Km, int ldk,
                       __hip_bfloat16* __restrict__ S,
                       const float* __restrict__ rel_emb,
                       const int* __restrict__ bucket1d,
                       const unsigned char* __restrict__ mask,
                       int Nq, int Nkv) {
    __shared__ bf16_t lsQ[128][64];
    __shared__ bf16_t lsK[128][64];
    int tid = threadIdx.x;
    int lane = tid & 31, wave = tid >> 5;
    int half = lane >> 4, l16 = lane & 15;
    int bh = blockIdx.z, b = bh >> 4, h = bh & 15;
    int i0 = blockIdx.y * 128, j0 = blockIdx.x * 128;
    for (int c = tid; c < 128 * 8; c += 256) {
        int r = c >> 3, q = c & 7;
        async_copy16(Q + (size_t)(b * Nq + i0 + r) * DMODEL + h * DHEAD + q * 8, &lsQ[r][q * 8]);
    }
    for (int c = tid; c < 128 * 8; c += 256) {
        int r = c >> 3, q = c & 7;
        async_copy16(Km + (size_t)(b * Nkv + j0 + r) * ldk + h * DHEAD + q * 8, &lsK[r][q * 8]);
    }
    wait_async_0();
    __syncthreads();
    int wr = (wave >> 1) * 32;
    int wc = (wave & 1) * 64;
    f32x8 acc[2][4];
#pragma unroll
    for (int i = 0; i < 2; ++i)
#pragma unroll
        for (int j = 0; j < 4; ++j) acc[i][j] = (f32x8)0.0f;
#pragma unroll
    for (int kk = 0; kk < 2; ++kk) {
        Frag a[2], bfr[4];
#pragma unroll
        for (int i = 0; i < 2; ++i)
            load_a_frag(a[i], (const unsigned*)&lsQ[wr + 16 * i + l16][0], kk * 16, half);
#pragma unroll
        for (int j = 0; j < 4; ++j)
            load_b_frag(bfr[j], (const unsigned*)&lsK[wc + 16 * j + l16][0], kk * 16, half);
#pragma unroll
        for (int i = 0; i < 2; ++i)
#pragma unroll
            for (int j = 0; j < 4; ++j) acc[i][j] = wmma_bf16(a[i], bfr[j], acc[i][j]);
    }
    const float scale = 0.125f;
#pragma unroll
    for (int i = 0; i < 2; ++i)
#pragma unroll
        for (int j = 0; j < 4; ++j) {
            int jj = j0 + wc + 16 * j + l16;
#pragma unroll
            for (int v = 0; v < 8; ++v) {
                int ii = i0 + wr + 16 * i + v + 8 * half;
                float sv = acc[i][j][v] * scale;
                if (rel_emb) sv += rel_emb[bucket1d[jj - ii + (Nq - 1)] * NHEAD + h];
                if (mask) {
                    bool ok = mask[b * Nq + ii] && mask[b * Nq + jj];
                    if (!ok) sv = -1e30f;
                }
                S[((size_t)bh * Nq + ii) * Nkv + jj] = __float2bfloat16(sv);
            }
        }
}

// ---------------- in-place row softmax over bf16 scores ----------------
__global__ __launch_bounds__(256)
void enc_softmax_kernel(__hip_bfloat16* __restrict__ S, int Nkv) {
    size_t row = blockIdx.x;
    __hip_bfloat16* p = S + row * (size_t)Nkv;
    int tid = threadIdx.x;
    float v[4]; int cnt = 0;
    float mx = -3.0e38f;
    for (int c = tid; c < Nkv; c += 256) {
        float f = __bfloat162float(p[c]);
        v[cnt++] = f; mx = fmaxf(mx, f);
    }
    __shared__ float red[8];
    for (int o = 16; o; o >>= 1) mx = fmaxf(mx, __shfl_xor(mx, o, 32));
    if ((tid & 31) == 0) red[tid >> 5] = mx;
    __syncthreads();
#pragma unroll
    for (int w = 0; w < 8; ++w) mx = fmaxf(mx, red[w]);
    float s = 0.f;
    for (int i = 0; i < cnt; ++i) { v[i] = __expf(v[i] - mx); s += v[i]; }
    __syncthreads();
    for (int o = 16; o; o >>= 1) s += __shfl_xor(s, o, 32);
    if ((tid & 31) == 0) red[tid >> 5] = s;
    __syncthreads();
    s = 0.f;
#pragma unroll
    for (int w = 0; w < 8; ++w) s += red[w];
    float inv = 1.0f / s;
    cnt = 0;
    for (int c = tid; c < Nkv; c += 256) p[c] = __float2bfloat16(v[cnt++] * inv);
}

// ---------------- O = P@V (both bf16) -> bf16 ----------------
__global__ __launch_bounds__(256)
void enc_pv_kernel(const __hip_bfloat16* __restrict__ P,
                   const bf16_t* __restrict__ V, int ldv,
                   bf16_t* __restrict__ O,
                   int Nq, int Nkv) {
    __shared__ bf16_t lsP[128][32];
    __shared__ unsigned lsV[64][16];
    int tid = threadIdx.x;
    int lane = tid & 31, wave = tid >> 5;
    int half = lane >> 4, l16 = lane & 15;
    int bh = blockIdx.z, b = bh >> 4, h = bh & 15;
    int i0 = blockIdx.y * 128;
    f32x8 acc[4];
#pragma unroll
    for (int j = 0; j < 4; ++j) acc[j] = (f32x8)0.0f;
    const unsigned short* Vs = (const unsigned short*)V;
    for (int j0 = 0; j0 < Nkv; j0 += 32) {
        __syncthreads();
        for (int c = tid; c < 128 * 2; c += 256) {
            int r = c >> 1, q = c & 1;
            async_copy16((const bf16_t*)P + ((size_t)bh * Nq + i0 + r) * Nkv + j0 + q * 8,
                         &lsP[r][q * 8]);
        }
        for (int e = tid; e < 64 * 16; e += 256) {
            int d = e & 63, p = e >> 6;
            unsigned lo = Vs[(size_t)(b * Nkv + j0 + 2 * p) * ldv + h * DHEAD + d];
            unsigned hi = Vs[(size_t)(b * Nkv + j0 + 2 * p + 1) * ldv + h * DHEAD + d];
            lsV[d][p] = lo | (hi << 16);
        }
        wait_async_0();
        __syncthreads();
        Frag a, bfr[4];
        load_a_frag(a, (const unsigned*)&lsP[wave * 16 + l16][0], 0, half);
#pragma unroll
        for (int j = 0; j < 4; ++j) load_b_frag(bfr[j], &lsV[16 * j + l16][0], 0, half);
#pragma unroll
        for (int j = 0; j < 4; ++j) acc[j] = wmma_bf16(a, bfr[j], acc[j]);
    }
#pragma unroll
    for (int j = 0; j < 4; ++j) {
        int col = 16 * j + l16;
#pragma unroll
        for (int v = 0; v < 8; ++v) {
            int ii = i0 + wave * 16 + v + 8 * half;
            O[(size_t)(b * Nq + ii) * DMODEL + h * DHEAD + col] = (bf16_t)acc[j][v];
        }
    }
}

// ---------------- copy ----------------
__global__ void enc_copy_kernel(const float* __restrict__ a, float* __restrict__ b, int n) {
    int i = blockIdx.x * 256 + threadIdx.x;
    if (i < n) b[i] = a[i];
}

// ---------------- host orchestration ----------------
extern "C" void kernel_launch(void* const* d_in, const int* in_sizes, int n_in,
                              void* d_out, int out_size, void* d_ws, size_t ws_size,
                              hipStream_t stream) {
    const float* x_in      = (const float*)d_in[0];
    const float* context   = (const float*)d_in[1];
    const unsigned char* src_mask = (const unsigned char*)d_in[2];
    const float* rel_emb   = (const float*)d_in[3];
    const float* sa_wq = (const float*)d_in[4];
    const float* sa_bq = (const float*)d_in[5];
    const float* sa_wkv = (const float*)d_in[6];
    const float* sa_bkv = (const float*)d_in[7];
    const float* sa_wo = (const float*)d_in[8];
    const float* sa_bo = (const float*)d_in[9];
    const float* sa_ln_g = (const float*)d_in[10];
    const float* sa_ln_b = (const float*)d_in[11];
    const float* ca_wq = (const float*)d_in[12];
    const float* ca_bq = (const float*)d_in[13];
    const float* ca_wkv = (const float*)d_in[14];
    const float* ca_bkv = (const float*)d_in[15];
    const float* ca_wo = (const float*)d_in[16];
    const float* ca_bo = (const float*)d_in[17];
    const float* ca_ln_g = (const float*)d_in[18];
    const float* ca_ln_b = (const float*)d_in[19];
    const float* ff_w1 = (const float*)d_in[20];
    const float* ff_b1 = (const float*)d_in[21];
    const float* ff_w2 = (const float*)d_in[22];
    const float* ff_b2 = (const float*)d_in[23];
    const float* ff_ln_g = (const float*)d_in[24];
    const float* ff_ln_b = (const float*)d_in[25];

    const int ROWS = BATCH * SEQN;            // 4096
    const int CROWS = BATCH * SEQM;           // 1024
    const int D = DMODEL;
    char* w = (char*)d_ws;
    float*  xbuf  = (float*)w;  w += (size_t)ROWS * D * 4;            // fp32 residual stream
    bf16_t* xn    = (bf16_t*)w; w += (size_t)ROWS * D * 2;
    bf16_t* qbuf  = (bf16_t*)w; w += (size_t)ROWS * D * 2;
    bf16_t* kvbuf = (bf16_t*)w; w += (size_t)ROWS * 2 * D * 2;
    bf16_t* aobuf = (bf16_t*)w; w += (size_t)ROWS * D * 2;
    bf16_t* hbuf  = (bf16_t*)w; w += (size_t)ROWS * 4 * D * 2;
    bf16_t* ctxb  = (bf16_t*)w; w += (size_t)CROWS * D * 2;           // bf16 context
    __hip_bfloat16* sbuf = (__hip_bfloat16*)w;
    w += (size_t)BATCH * NHEAD * SEQN * SEQN * 2;
    int* bucket = (int*)w;      w += 16384;
    bf16_t* wtbase = (bf16_t*)w;                                      // bf16 W^T, 16M elems/layer
    const size_t PER = 16u * 1024 * 1024;
    const size_t M1 = 1024u * 1024;
    auto WT = [&](int l, size_t off) { return wtbase + (size_t)l * PER + off; };

    dim3 blk(256);
    for (int l = 0; l < NLAYER; ++l) {
        enc_convw_kernel<<<dim3(D / 32, D / 32), blk, 0, stream>>>(
            sa_wq + (size_t)l * D * D, D, D, WT(l, 0));
        enc_convw_kernel<<<dim3(2 * D / 32, D / 32), blk, 0, stream>>>(
            sa_wkv + (size_t)l * D * 2 * D, D, 2 * D, WT(l, M1));
        enc_convw_kernel<<<dim3(D / 32, D / 32), blk, 0, stream>>>(
            sa_wo + (size_t)l * D * D, D, D, WT(l, 3 * M1));
        enc_convw_kernel<<<dim3(D / 32, D / 32), blk, 0, stream>>>(
            ca_wq + (size_t)l * D * D, D, D, WT(l, 4 * M1));
        enc_convw_kernel<<<dim3(2 * D / 32, D / 32), blk, 0, stream>>>(
            ca_wkv + (size_t)l * D * 2 * D, D, 2 * D, WT(l, 5 * M1));
        enc_convw_kernel<<<dim3(D / 32, D / 32), blk, 0, stream>>>(
            ca_wo + (size_t)l * D * D, D, D, WT(l, 7 * M1));
        enc_convw_kernel<<<dim3(4 * D / 32, D / 32), blk, 0, stream>>>(
            ff_w1 + (size_t)l * D * 4 * D, D, 4 * D, WT(l, 8 * M1));
        enc_convw_kernel<<<dim3(D / 32, 4 * D / 32), blk, 0, stream>>>(
            ff_w2 + (size_t)l * 4 * D * D, 4 * D, D, WT(l, 12 * M1));
    }
    enc_bucket_kernel<<<dim3((2 * SEQN - 1 + 255) / 256), blk, 0, stream>>>(bucket, SEQN, SEQN);
    enc_copy_kernel<<<dim3(ROWS * D / 256), blk, 0, stream>>>(x_in, xbuf, ROWS * D);
    enc_convx_kernel<<<dim3(CROWS * D / 256), blk, 0, stream>>>(context, ctxb, CROWS * D);

    for (int l = 0; l < NLAYER; ++l) {
        // ---- self attention ----
        enc_layernorm_kernel<<<dim3(ROWS), blk, 0, stream>>>(
            xbuf, sa_ln_g + (size_t)l * D, sa_ln_b + (size_t)l * D, xn);
        enc_gemm_kernel<<<dim3(D / 128, ROWS / 128), blk, 0, stream>>>(
            xn, D, WT(l, 0), D, sa_bq + (size_t)l * D, nullptr, nullptr, qbuf, D, D, 0);
        enc_gemm_kernel<<<dim3(2 * D / 128, ROWS / 128), blk, 0, stream>>>(
            xn, D, WT(l, M1), D, sa_bkv + (size_t)l * 2 * D, nullptr, nullptr, kvbuf, 2 * D, D, 0);
        enc_scores_kernel<<<dim3(SEQN / 128, SEQN / 128, BATCH * NHEAD), blk, 0, stream>>>(
            qbuf, kvbuf, 2 * D, sbuf, rel_emb, bucket, src_mask, SEQN, SEQN);
        enc_softmax_kernel<<<dim3(BATCH * NHEAD * SEQN), blk, 0, stream>>>(sbuf, SEQN);
        enc_pv_kernel<<<dim3(1, SEQN / 128, BATCH * NHEAD), blk, 0, stream>>>(
            sbuf, kvbuf + D, 2 * D, aobuf, SEQN, SEQN);
        enc_gemm_kernel<<<dim3(D / 128, ROWS / 128), blk, 0, stream>>>(
            aobuf, D, WT(l, 3 * M1), D, sa_bo + (size_t)l * D, xbuf, xbuf, nullptr, D, D, 0);

        // ---- cross attention ----
        enc_layernorm_kernel<<<dim3(ROWS), blk, 0, stream>>>(
            xbuf, ca_ln_g + (size_t)l * D, ca_ln_b + (size_t)l * D, xn);
        enc_gemm_kernel<<<dim3(D / 128, ROWS / 128), blk, 0, stream>>>(
            xn, D, WT(l, 4 * M1), D, ca_bq + (size_t)l * D, nullptr, nullptr, qbuf, D, D, 0);
        enc_gemm_kernel<<<dim3(2 * D / 128, CROWS / 128), blk, 0, stream>>>(
            ctxb, D, WT(l, 5 * M1), D, ca_bkv + (size_t)l * 2 * D, nullptr, nullptr, kvbuf, 2 * D, D, 0);
        enc_scores_kernel<<<dim3(SEQM / 128, SEQN / 128, BATCH * NHEAD), blk, 0, stream>>>(
            qbuf, kvbuf, 2 * D, sbuf, nullptr, nullptr, nullptr, SEQN, SEQM);
        enc_softmax_kernel<<<dim3(BATCH * NHEAD * SEQN), blk, 0, stream>>>(sbuf, SEQM);
        enc_pv_kernel<<<dim3(1, SEQN / 128, BATCH * NHEAD), blk, 0, stream>>>(
            sbuf, kvbuf + D, 2 * D, aobuf, SEQN, SEQM);
        enc_gemm_kernel<<<dim3(D / 128, ROWS / 128), blk, 0, stream>>>(
            aobuf, D, WT(l, 7 * M1), D, ca_bo + (size_t)l * D, xbuf, xbuf, nullptr, D, D, 0);

        // ---- feed forward ----
        enc_layernorm_kernel<<<dim3(ROWS), blk, 0, stream>>>(
            xbuf, ff_ln_g + (size_t)l * D, ff_ln_b + (size_t)l * D, xn);
        enc_gemm_kernel<<<dim3(4 * D / 128, ROWS / 128), blk, 0, stream>>>(
            xn, D, WT(l, 8 * M1), D, ff_b1 + (size_t)l * 4 * D, nullptr, nullptr, hbuf, 4 * D, D, 1);
        enc_gemm_kernel<<<dim3(D / 128, ROWS / 128), blk, 0, stream>>>(
            hbuf, 4 * D, WT(l, 12 * M1), 4 * D, ff_b2 + (size_t)l * D, xbuf, xbuf, nullptr, D, 4 * D, 0);
    }
    enc_copy_kernel<<<dim3(ROWS * D / 256), blk, 0, stream>>>(xbuf, (float*)d_out, ROWS * D);

    (void)in_sizes; (void)n_in; (void)out_size; (void)ws_size;
}
